// GraphSage_network_22316650070763
// MI455X (gfx1250) — matrix operand order repaired and checked by reading, not compiled, not run on hardware
//
#include <hip/hip_runtime.h>
#include <hip/hip_bf16.h>

// ---------------------------------------------------------------------------
// GraphSage GNN forward for MI455X (gfx1250, wave32, WMMA).
// Dominant compute (edge MLP 16->256->128 over 800k edges) runs on
// v_wmma_f32_16x16x32_bf16; scatters use global float atomics (L2 resident).
// Tiny decoder (G=1024) stays fp32 VALU for exact numerics.
// ---------------------------------------------------------------------------

typedef __bf16 bf16_t;
typedef __attribute__((ext_vector_type(16))) __bf16 v16bf;
typedef __attribute__((ext_vector_type(8)))  __bf16 v8bf;
typedef __attribute__((ext_vector_type(8)))  float  v8f;

#define NN 100000
#define NE 800000
#define NG 1024

static __device__ __forceinline__ v8f wmma_bf16(v16bf a, v16bf b, v8f c) {
  return __builtin_amdgcn_wmma_f32_16x16x32_bf16(false, a, false, b, (short)0, c, false, false);
}

// A-fragment (16x32 bf16, M x K) from row-major bf16 [16][stride] in LDS.
// lane<16: K = {0..7, 16..23}; lane>=16: K = {8..15, 24..31} (ISA 7.12.2).
static __device__ __forceinline__ v16bf frag_a_lds(const bf16_t* base, int stride,
                                                   int kblk, int lane) {
  int r  = lane & 15;
  int kb = (lane & 16) ? 8 : 0;
  const bf16_t* p = base + r * stride + kblk * 32 + kb;
  v8bf lo = *(const v8bf*)(p);
  v8bf hi = *(const v8bf*)(p + 16);
  v16bf f;
#pragma unroll
  for (int i = 0; i < 8; ++i) { f[i] = lo[i]; f[i + 8] = hi[i]; }
  return f;
}

// B-fragment (32x16 bf16, K x N) from weights stored transposed [Nout][K].
// lane<16: K = {0..7, 16..23}; lane>=16: K = {8..15, 24..31} (matches A).
static __device__ __forceinline__ v16bf frag_b_gl(const bf16_t* wt, int K,
                                                  int nrow, int kblk, int lane) {
  const bf16_t* p = wt + (size_t)nrow * K + kblk * 32 + ((lane & 16) ? 8 : 0);
  v8bf lo = *(const v8bf*)(p);
  v8bf hi = *(const v8bf*)(p + 16);
  v16bf f;
#pragma unroll
  for (int i = 0; i < 8; ++i) { f[i] = lo[i]; f[i + 8] = hi[i]; }
  return f;
}

// ------------------------------- utilities ---------------------------------

__global__ void k_zero(float* p, long n) {
  long i  = (long)blockIdx.x * blockDim.x + threadIdx.x;
  long st = (long)gridDim.x * blockDim.x;
  for (; i < n; i += st) p[i] = 0.0f;
}

// w: [Kin][Nout] fp32 -> wt: [Nout][Kpad] bf16 (zero-padded K)
__global__ void k_transpose_bf16(const float* __restrict__ w, bf16_t* __restrict__ wt,
                                 int Kin, int Nout, int Kpad) {
  int idx = blockIdx.x * blockDim.x + threadIdx.x;
  if (idx >= Nout * Kpad) return;
  int n = idx / Kpad, k = idx % Kpad;
  wt[idx] = (k < Kin) ? (bf16_t)w[(size_t)k * Nout + n] : (bf16_t)0.0f;
}

__global__ void k_degree(const int* __restrict__ ea, float* __restrict__ deg) {
  int e = blockIdx.x * blockDim.x + threadIdx.x;
  if (e < NE) atomicAdd(&deg[ea[2 * e + 1]], 1.0f);
}

// --------------------------- edge message MLP ------------------------------
// 4 waves/block, each wave: 16 edges. feat(16) -> 256 relu -> 128 relu,
// then atomic segment-sum into hsum[recv].
__global__ void __launch_bounds__(128) k_edge_mlp(
    const float* __restrict__ x, const int* __restrict__ ea,
    const float* __restrict__ bnS, const float* __restrict__ bnT,
    const bf16_t* __restrict__ wt1, const float* __restrict__ b1,
    const bf16_t* __restrict__ wt2, const float* __restrict__ b2,
    float* __restrict__ hsum) {
  __shared__ __align__(16) bf16_t sm_m[4][16][32];
  __shared__ __align__(16) bf16_t sm_y[4][16][256];
  __shared__ int sm_recv[4][16];
  const int lane = threadIdx.x & 31;
  const int w    = threadIdx.x >> 5;
  const int tile = blockIdx.x * 4 + w;
  const int eb   = tile * 16;

  if (lane < 16) {
    const int e = eb + lane;
    const int s = ea[2 * e], r = ea[2 * e + 1];
    sm_recv[w][lane] = r;
    const float TR[5] = {0.f, 0.f, -200.f, 10000.f, 0.f};
    const float SC[5] = {100.f, 100.f, 100.f, 2500.f, 0.25f};
    float xs[5], xr[5], df[5];
#pragma unroll
    for (int d = 0; d < 5; ++d) {
      xs[d] = (x[(size_t)s * 5 + d] - TR[d]) / SC[d];
      xr[d] = (x[(size_t)r * 5 + d] - TR[d]) / SC[d];
      df[d] = xr[d] - xs[d];
    }
    float d2   = df[0] * df[0] + df[1] * df[1] + df[2] * df[2];
    float dist = sqrtf(fmaxf(d2, 1e-24f));
    float inv  = 1.0f / dist;
    float ef[6] = {df[3], df[4], dist, df[0] * inv, df[1] * inv, df[2] * inv};
    float m[16];
#pragma unroll
    for (int d = 0; d < 5; ++d) { m[d] = xr[d]; m[5 + d] = xs[d]; }
#pragma unroll
    for (int j = 0; j < 6; ++j) m[10 + j] = ef[j] * bnS[j] + bnT[j];
#pragma unroll
    for (int k = 0; k < 16; ++k) sm_m[w][lane][k] = (bf16_t)m[k];
#pragma unroll
    for (int k = 16; k < 32; ++k) sm_m[w][lane][k] = (bf16_t)0.0f;
  }
  __syncthreads();

  const int col0 = lane & 15;
  const int roff = (lane & 16) ? 8 : 0;

  // layer1: 16 -> 256 (K padded to 32)
  v16bf af = frag_a_lds(&sm_m[w][0][0], 32, 0, lane);
  for (int ct = 0; ct < 16; ++ct) {
    v16bf bf = frag_b_gl(wt1, 32, ct * 16 + col0, 0, lane);
    v8f c = {};
    c = wmma_bf16(af, bf, c);
    const int col  = ct * 16 + col0;
    const float bb = b1[col];
#pragma unroll
    for (int r = 0; r < 8; ++r)
      sm_y[w][roff + r][col] = (bf16_t)fmaxf(c[r] + bb, 0.0f);
  }
  __syncthreads();

  // layer2: 256 -> 128, fused atomic segment-sum
  v8f acc[8];
#pragma unroll
  for (int ct = 0; ct < 8; ++ct) acc[ct] = (v8f){};
  for (int kb = 0; kb < 8; ++kb) {
    v16bf a2 = frag_a_lds(&sm_y[w][0][0], 256, kb, lane);
#pragma unroll
    for (int ct = 0; ct < 8; ++ct) {
      v16bf bf = frag_b_gl(wt2, 256, ct * 16 + col0, kb, lane);
      acc[ct] = wmma_bf16(a2, bf, acc[ct]);
    }
  }
#pragma unroll
  for (int ct = 0; ct < 8; ++ct) {
    const int col  = ct * 16 + col0;
    const float bb = b2[col];
#pragma unroll
    for (int r = 0; r < 8; ++r) {
      float v = fmaxf(acc[ct][r] + bb, 0.0f);
      atomicAdd(&hsum[(size_t)sm_recv[w][roff + r] * 128 + col], v);
    }
  }
}

// ------------------------- node update MLP (128->128->64) ------------------
__global__ void __launch_bounds__(64) k_node_update(
    const float* __restrict__ hin, const bf16_t* __restrict__ wtu1,
    const float* __restrict__ bu1, const bf16_t* __restrict__ wtu2,
    const float* __restrict__ bu2, float* __restrict__ hout) {
  __shared__ __align__(16) bf16_t at[2][16][128];
  __shared__ __align__(16) bf16_t yt[2][16][128];
  const int lane = threadIdx.x & 31;
  const int w    = threadIdx.x >> 5;
  const int base = (blockIdx.x * 2 + w) * 16;

  for (int idx = lane; idx < 16 * 128; idx += 32) {
    int r = idx >> 7, c = idx & 127;
    at[w][r][c] = (bf16_t)hin[(size_t)(base + r) * 128 + c];
  }
  __syncthreads();

  const int col0 = lane & 15;
  const int roff = (lane & 16) ? 8 : 0;

  v8f acc[8];
#pragma unroll
  for (int ct = 0; ct < 8; ++ct) acc[ct] = (v8f){};
  for (int kb = 0; kb < 4; ++kb) {
    v16bf af = frag_a_lds(&at[w][0][0], 128, kb, lane);
#pragma unroll
    for (int ct = 0; ct < 8; ++ct)
      acc[ct] = wmma_bf16(af, frag_b_gl(wtu1, 128, ct * 16 + col0, kb, lane), acc[ct]);
  }
#pragma unroll
  for (int ct = 0; ct < 8; ++ct) {
    const int col = ct * 16 + col0;
    const float bb = bu1[col];
#pragma unroll
    for (int r = 0; r < 8; ++r)
      yt[w][roff + r][col] = (bf16_t)fmaxf(acc[ct][r] + bb, 0.0f);
  }
  __syncthreads();

  v8f a2[4];
#pragma unroll
  for (int ct = 0; ct < 4; ++ct) a2[ct] = (v8f){};
  for (int kb = 0; kb < 4; ++kb) {
    v16bf af = frag_a_lds(&yt[w][0][0], 128, kb, lane);
#pragma unroll
    for (int ct = 0; ct < 4; ++ct)
      a2[ct] = wmma_bf16(af, frag_b_gl(wtu2, 128, ct * 16 + col0, kb, lane), a2[ct]);
  }
#pragma unroll
  for (int ct = 0; ct < 4; ++ct) {
    const int col = ct * 16 + col0;
    const float bb = bu2[col];
#pragma unroll
    for (int r = 0; r < 8; ++r)
      hout[(size_t)(base + roff + r) * 64 + col] = fmaxf(a2[ct][r] + bb, 0.0f);
  }
}

// ------------------------------ scatter h[send] ----------------------------
// One thread handles an 8-column chunk of one edge: indices loaded once per
// chunk, data via float4, 8 irreducible atomics.
template <int DIM>
__global__ void k_scatter(const int* __restrict__ ea, const float* __restrict__ h,
                          float* __restrict__ nb) {
  constexpr int CH  = DIM / 8;                    // chunks per edge (8 or 16)
  constexpr int CSH = (DIM == 64) ? 3 : 4;        // log2(CH)
  long idx = (long)blockIdx.x * blockDim.x + threadIdx.x;
  if (idx >= (long)NE * CH) return;
  long e  = idx >> CSH;
  int  ch = (int)(idx & (CH - 1));
  int s = ea[2 * e], r = ea[2 * e + 1];
  const float4* src = (const float4*)(h + (size_t)s * DIM + ch * 8);
  float4 v0 = src[0], v1 = src[1];
  float* dst = nb + (size_t)r * DIM + ch * 8;
  atomicAdd(dst + 0, v0.x); atomicAdd(dst + 1, v0.y);
  atomicAdd(dst + 2, v0.z); atomicAdd(dst + 3, v0.w);
  atomicAdd(dst + 4, v1.x); atomicAdd(dst + 5, v1.y);
  atomicAdd(dst + 6, v1.z); atomicAdd(dst + 7, v1.w);
}

// ------------------------ SAGE layer: concat->GEMM->l2norm->relu -----------
template <int DIN>
__global__ void __launch_bounds__(64) k_sage(
    const float* __restrict__ hprev, const float* __restrict__ nb,
    const float* __restrict__ deg, const bf16_t* __restrict__ wt,
    const float* __restrict__ b, float* __restrict__ hout) {
  constexpr int HALF = DIN / 2;
  constexpr int SH   = (DIN == 128) ? 7 : 8;
  constexpr int NCT  = DIN / 16;
  constexpr int NKB  = DIN / 32;
  __shared__ __align__(16) bf16_t at[2][16][DIN];
  __shared__ __align__(16) float  zt[2][16][DIN];
  __shared__ float sscale[2][16];
  const int lane = threadIdx.x & 31;
  const int w    = threadIdx.x >> 5;
  const int base = (blockIdx.x * 2 + w) * 16;

  for (int idx = lane; idx < 16 * DIN; idx += 32) {
    int r = idx >> SH, c = idx & (DIN - 1);
    int node = base + r;
    float v;
    if (c < HALF) {
      v = hprev[(size_t)node * HALF + c];
    } else {
      float d = deg[node];
      float s = nb[(size_t)node * HALF + (c - HALF)];
      v = (d > 0.0f) ? s / fmaxf(d, 1.0f) : 0.0f;
    }
    at[w][r][c] = (bf16_t)v;
  }
  __syncthreads();

  const int col0 = lane & 15;
  const int roff = (lane & 16) ? 8 : 0;

  v8f acc[NCT];
#pragma unroll
  for (int ct = 0; ct < NCT; ++ct) acc[ct] = (v8f){};
  for (int kb = 0; kb < NKB; ++kb) {
    v16bf af = frag_a_lds(&at[w][0][0], DIN, kb, lane);   // hoisted per k-block
#pragma unroll
    for (int ct = 0; ct < NCT; ++ct) {
      v16bf bf = frag_b_gl(wt, DIN, ct * 16 + col0, kb, lane);
      acc[ct] = wmma_bf16(af, bf, acc[ct]);
    }
  }
#pragma unroll
  for (int ct = 0; ct < NCT; ++ct) {
    const int col = ct * 16 + col0;
    const float bb = b[col];
#pragma unroll
    for (int r = 0; r < 8; ++r) zt[w][roff + r][col] = acc[ct][r] + bb;
  }
  __syncthreads();

  if (lane < 16) {
    float ss = 0.0f;
    for (int c = 0; c < DIN; ++c) { float z = zt[w][lane][c]; ss += z * z; }
    sscale[w][lane] = rsqrtf(fmaxf(ss, 1e-12f));
  }
  __syncthreads();

  for (int idx = lane; idx < 16 * DIN; idx += 32) {
    int r = idx >> SH, c = idx & (DIN - 1);
    hout[(size_t)(base + r) * DIN + c] = fmaxf(zt[w][r][c] * sscale[w][r], 0.0f);
  }
}

// ------------------------------ group reductions ---------------------------
__global__ void k_group_cnt(const int* __restrict__ gi, float* __restrict__ gcnt) {
  int n = blockIdx.x * blockDim.x + threadIdx.x;
  if (n < NN) atomicAdd(&gcnt[gi[n]], 1.0f);
}

// Thread per (node, 8-col chunk): group id loaded once per chunk.
__global__ void k_group_stats(const float* __restrict__ h, const int* __restrict__ gi,
                              float* __restrict__ gsum, float* __restrict__ gmx) {
  long idx = (long)blockIdx.x * blockDim.x + threadIdx.x;
  if (idx >= (long)NN * 32) return;
  int node = (int)(idx >> 5);
  int ch   = (int)(idx & 31);
  int g    = gi[node];
  const float4* src = (const float4*)(h + (size_t)node * 256 + ch * 8);
  float4 a = src[0], bq = src[1];
  float vals[8] = {a.x, a.y, a.z, a.w, bq.x, bq.y, bq.z, bq.w};
  size_t base = (size_t)g * 256 + ch * 8;
#pragma unroll
  for (int j = 0; j < 8; ++j) {
    atomicAdd(&gsum[base + j], vals[j]);
    // h >= 0 (post-relu): int bit pattern monotone; init 0 matches the
    // reference's "empty segment -> 0" replacement.
    atomicMax((int*)&gmx[base + j], __float_as_int(vals[j]));
  }
}

__global__ void k_compose(const float* __restrict__ gmx, const float* __restrict__ gsum,
                          const float* __restrict__ gcnt, float* __restrict__ gbuf) {
  int idx = blockIdx.x * blockDim.x + threadIdx.x;
  if (idx >= NG * 768) return;
  int g = idx / 768, c = idx % 768;
  float v;
  if (c < 256) {
    v = gmx[(size_t)g * 256 + c];
  } else if (c < 512) {
    float cnt = gcnt[g];
    float s = gsum[(size_t)g * 256 + (c - 256)];
    v = (cnt > 0.0f) ? s / fmaxf(cnt, 1.0f) : 0.0f;
  } else {
    v = gsum[(size_t)g * 256 + (c - 512)];
  }
  gbuf[idx] = v;
}

// ------------------------------- decoder (fp32) ----------------------------
struct DecParams {
  const float *w1, *b1, *bn1s, *bn1t;
  const float *w2, *b2, *bn2s, *bn2t;
  const float *w3, *b3, *bn3s, *bn3t;
  const float *wu1, *bu1, *wu2, *bu2, *wuo, *buo;
  const float *ws1, *bs1, *ws2, *bs2, *wso, *bso;
};

__global__ void __launch_bounds__(384) k_decoder(const float* __restrict__ gbuf,
                                                 DecParams p, float* __restrict__ out) {
  __shared__ float gin[768];
  __shared__ float L1[384];
  __shared__ float L2[384];
  __shared__ float L3[192];
  __shared__ float hu[64], hu2[64], hs[64], hs2[64];
  __shared__ float uo[3];
  const int g = blockIdx.x;
  const int t = threadIdx.x;
  for (int c = t; c < 768; c += 384) gin[c] = gbuf[(size_t)g * 768 + c];
  __syncthreads();
  {
    float a = p.b1[t];
    for (int k = 0; k < 768; ++k) a += gin[k] * p.w1[(size_t)k * 384 + t];
    float v = (a > 0.0f) ? a : 0.15f * a;
    L1[t] = v * p.bn1s[t] + p.bn1t[t];
  }
  __syncthreads();
  {
    float a = p.b2[t];
    for (int k = 0; k < 384; ++k) a += L1[k] * p.w2[(size_t)k * 384 + t];
    float v = (a > 0.0f) ? a : 0.15f * a;
    L2[t] = v * p.bn2s[t] + p.bn2t[t];
  }
  __syncthreads();
  if (t < 192) {
    float a = p.b3[t];
    for (int k = 0; k < 384; ++k) a += L2[k] * p.w3[(size_t)k * 192 + t];
    float v = (a > 0.0f) ? a : 0.15f * a;
    L3[t] = v * p.bn3s[t] + p.bn3t[t];
  }
  __syncthreads();
  if (t < 64) {
    float a1 = p.bu1[t], a2 = p.bs1[t];
    for (int k = 0; k < 192; ++k) {
      float gk = L3[k];
      a1 += gk * p.wu1[(size_t)k * 64 + t];
      a2 += gk * p.ws1[(size_t)k * 64 + t];
    }
    hu[t] = a1; hs[t] = a2;
  }
  __syncthreads();
  if (t < 64) {
    float a1 = p.bu2[t], a2 = p.bs2[t];
    for (int k = 0; k < 64; ++k) {
      a1 += hu[k] * p.wu2[(size_t)k * 64 + t];
      a2 += hs[k] * p.ws2[(size_t)k * 64 + t];
    }
    hu2[t] = a1; hs2[t] = a2;
  }
  __syncthreads();
  if (t < 3) {
    float a = p.buo[t];
    for (int k = 0; k < 64; ++k) a += hu2[k] * p.wuo[(size_t)k * 3 + t];
    uo[t] = a;
  }
  __syncthreads();
  if (t == 0) {
    float un = sqrtf(fmaxf(uo[0] * uo[0] + uo[1] * uo[1] + uo[2] * uo[2], 1e-30f));
    out[(size_t)g * 4 + 0] = uo[0] / un;
    out[(size_t)g * 4 + 1] = uo[1] / un;
    out[(size_t)g * 4 + 2] = uo[2] / un;
    float sa = p.bso[0];
    for (int k = 0; k < 64; ++k) sa += hs2[k] * p.wso[k];
    out[(size_t)g * 4 + 3] = fabsf(sa) + 1e-5f;
  }
}

// ------------------------------- launcher ----------------------------------
static inline int cdiv(long a, long b) { return (int)((a + b - 1) / b); }

extern "C" void kernel_launch(void* const* d_in, const int* in_sizes, int n_in,
                              void* d_out, int out_size, void* d_ws, size_t ws_size,
                              hipStream_t stream) {
  (void)in_sizes; (void)n_in; (void)out_size; (void)ws_size;

  const float* x      = (const float*)d_in[0];
  const int*   ea     = (const int*)d_in[1];
  const int*   gi     = (const int*)d_in[2];
  const float* bnS    = (const float*)d_in[3];
  const float* bnT    = (const float*)d_in[4];
  const float* w_msg1 = (const float*)d_in[5];
  const float* b_msg1 = (const float*)d_in[6];
  const float* w_msg2 = (const float*)d_in[7];
  const float* b_msg2 = (const float*)d_in[8];
  const float* w_upd1 = (const float*)d_in[9];
  const float* b_upd1 = (const float*)d_in[10];
  const float* w_upd2 = (const float*)d_in[11];
  const float* b_upd2 = (const float*)d_in[12];
  const float* w_sg1  = (const float*)d_in[13];
  const float* b_sg1  = (const float*)d_in[14];
  const float* w_sg2  = (const float*)d_in[15];
  const float* b_sg2  = (const float*)d_in[16];

  DecParams p;
  p.w1 = (const float*)d_in[17]; p.b1 = (const float*)d_in[18];
  p.bn1s = (const float*)d_in[19]; p.bn1t = (const float*)d_in[20];
  p.w2 = (const float*)d_in[21]; p.b2 = (const float*)d_in[22];
  p.bn2s = (const float*)d_in[23]; p.bn2t = (const float*)d_in[24];
  p.w3 = (const float*)d_in[25]; p.b3 = (const float*)d_in[26];
  p.bn3s = (const float*)d_in[27]; p.bn3t = (const float*)d_in[28];
  p.wu1 = (const float*)d_in[29]; p.bu1 = (const float*)d_in[30];
  p.wu2 = (const float*)d_in[31]; p.bu2 = (const float*)d_in[32];
  p.wuo = (const float*)d_in[33]; p.buo = (const float*)d_in[34];
  p.ws1 = (const float*)d_in[35]; p.bs1 = (const float*)d_in[36];
  p.ws2 = (const float*)d_in[37]; p.bs2 = (const float*)d_in[38];
  p.wso = (const float*)d_in[39]; p.bso = (const float*)d_in[40];

  float* out = (float*)d_out;

  // ---- workspace layout (fp32 unless noted) ----
  float* F = (float*)d_ws;
  size_t o = 0;
  float* hsum = F + o; o += (size_t)NN * 128;     // regA: hsum, then nb1/nb2
  float* h128 = F + o; o += (size_t)NN * 128;     // regB
  float* h256 = F + o; float* h64 = h256; o += (size_t)NN * 256;  // regC (h64 aliases, dead before h256 written)
  float* deg  = F + o; o += NN;
  float* gsum = F + o; o += (size_t)NG * 256;
  float* gmx  = F + o; o += (size_t)NG * 256;
  float* gcnt = F + o; o += NG;
  float* gbuf = F + o; o += (size_t)NG * 768;
  bf16_t* wb   = (bf16_t*)(F + o);
  bf16_t* wt1  = wb;                   // [256][32]
  bf16_t* wt2  = wt1 + 256 * 32;       // [128][256]
  bf16_t* wtu1 = wt2 + 128 * 256;      // [128][128]
  bf16_t* wtu2 = wtu1 + 128 * 128;     // [64][128]
  bf16_t* wts1 = wtu2 + 64 * 128;      // [128][128]
  bf16_t* wts2 = wts1 + 128 * 128;     // [256][256]
  float* nb1 = hsum;   // after node_update, hsum region is free
  float* nb2 = hsum;

  // ---- zero accumulators ----
  k_zero<<<cdiv((long)NN * 128, 256), 256, 0, stream>>>(hsum, (long)NN * 128);
  k_zero<<<cdiv(NN, 256), 256, 0, stream>>>(deg, NN);
  k_zero<<<cdiv((long)NG * 256, 256), 256, 0, stream>>>(gsum, (long)NG * 256);
  k_zero<<<cdiv((long)NG * 256, 256), 256, 0, stream>>>(gmx, (long)NG * 256);
  k_zero<<<cdiv(NG, 256), 256, 0, stream>>>(gcnt, NG);

  // ---- prep bf16 transposed weights ----
  k_transpose_bf16<<<cdiv(256 * 32, 256), 256, 0, stream>>>(w_msg1, wt1, 16, 256, 32);
  k_transpose_bf16<<<cdiv(128 * 256, 256), 256, 0, stream>>>(w_msg2, wt2, 256, 128, 256);
  k_transpose_bf16<<<cdiv(128 * 128, 256), 256, 0, stream>>>(w_upd1, wtu1, 128, 128, 128);
  k_transpose_bf16<<<cdiv(64 * 128, 256), 256, 0, stream>>>(w_upd2, wtu2, 128, 64, 128);
  k_transpose_bf16<<<cdiv(128 * 128, 256), 256, 0, stream>>>(w_sg1, wts1, 128, 128, 128);
  k_transpose_bf16<<<cdiv(256 * 256, 256), 256, 0, stream>>>(w_sg2, wts2, 256, 256, 256);

  // ---- degree ----
  k_degree<<<cdiv(NE, 256), 256, 0, stream>>>(ea, deg);

  // ---- edge MLP + segment sum (WMMA) ----
  k_edge_mlp<<<NE / 64, 128, 0, stream>>>(x, ea, bnS, bnT, wt1, b_msg1, wt2, b_msg2, hsum);

  // ---- node update MLP (WMMA) ----
  k_node_update<<<NN / 32, 64, 0, stream>>>(hsum, wtu1, b_upd1, wtu2, b_upd2, h64);

  // ---- SAGE layer 1 (64 -> 128) ----
  k_zero<<<cdiv((long)NN * 64, 256), 256, 0, stream>>>(nb1, (long)NN * 64);
  k_scatter<64><<<cdiv((long)NE * 8, 256), 256, 0, stream>>>(ea, h64, nb1);
  k_sage<128><<<NN / 32, 64, 0, stream>>>(h64, nb1, deg, wts1, b_sg1, h128);

  // ---- SAGE layer 2 (128 -> 256) ----
  k_zero<<<cdiv((long)NN * 128, 256), 256, 0, stream>>>(nb2, (long)NN * 128);
  k_scatter<128><<<cdiv((long)NE * 16, 256), 256, 0, stream>>>(ea, h128, nb2);
  k_sage<256><<<NN / 32, 64, 0, stream>>>(h128, nb2, deg, wts2, b_sg2, h256);

  // ---- group pooling ----
  k_group_cnt<<<cdiv(NN, 256), 256, 0, stream>>>(gi, gcnt);
  k_group_stats<<<cdiv((long)NN * 32, 256), 256, 0, stream>>>(h256, gi, gsum, gmx);
  k_compose<<<cdiv((long)NG * 768, 256), 256, 0, stream>>>(gmx, gsum, gcnt, gbuf);

  // ---- decoder + heads (fp32) ----
  k_decoder<<<NG, 384, 0, stream>>>(gbuf, p, out);
}
